// KalmanNet_17179869899
// MI455X (gfx1250) — matrix operand order repaired and checked by reading, not compiled, not run on hardware
//
#include <hip/hip_runtime.h>
#include <cstdint>

// ---------------------------------------------------------------------------
// KalmanNet forward, MI455X (gfx1250, wave32, WMMA).
// All layers run as batched GEMMs C[B,N] = act(A[B,K] @ W[N,K]^T + b) on
// v_wmma_f32_16x16x32_f16 (f16 operands, f32 accumulate).
// All operands are padded (K to mult-of-32, N to mult-of-16, zero-filled) so
// every inner-loop load is an unconditional 32-byte vector load — no per-lane
// predication. fc2a->relu->fc2b (and fc21*) are fused per 16-row batch tile:
// the 3400-wide hidden lives only in a 16x32 LDS tile (C-frag -> A-frag).
// ---------------------------------------------------------------------------

typedef __attribute__((ext_vector_type(16))) _Float16 v16h;
typedef __attribute__((ext_vector_type(8)))  float    v8f;

#define BATCH 16384
#define DT_C 0.01f

__device__ __forceinline__ v8f wmma_f16(v16h a, v16h b, v8f c) {
  return __builtin_amdgcn_wmma_f32_16x16x32_f16(
      false, a, false, b, (short)0, c, false, false);
}

// ---------------- zero the f32 activation region ----------------------------
__global__ void zero_ws_kernel(float4* __restrict__ p, size_t n4) {
  size_t i = (size_t)blockIdx.x * 256 + threadIdx.x;
  if (i < n4) p[i] = make_float4(0.f, 0.f, 0.f, 0.f);
}

// ---------------- f32 -> f16 weight conversion with [Np,Kp] zero padding ----
__global__ void cvt_pad_w(const float* __restrict__ src, _Float16* __restrict__ dst,
                          int N, int K, int Kp, int total) {
  int i = blockIdx.x * 256 + threadIdx.x;
  if (i >= total) return;
  int n = i / Kp, k = i - n * Kp;
  dst[i] = (n < N && k < K) ? (_Float16)src[n * K + k] : (_Float16)0.0f;
}

// ---------------- bias copy with zero padding -------------------------------
__global__ void pad_bias(const float* __restrict__ src, float* __restrict__ dst,
                         int N, int Np) {
  int i = blockIdx.x * 256 + threadIdx.x;
  if (i < Np) dst[i] = (i < N) ? src[i] : 0.0f;
}

// ---------------- pack small inputs into padded activation buffers ----------
__global__ void pack_inputs(const float* __restrict__ fwe, const float* __restrict__ fwu,
                            const float* __restrict__ hq, const float* __restrict__ hsig,
                            const float* __restrict__ hs,
                            float* __restrict__ pfe, float* __restrict__ pfu,
                            float* __restrict__ phq, float* __restrict__ phsig,
                            float* __restrict__ phs) {
  int i = blockIdx.x * 256 + threadIdx.x;
  if (i >= BATCH) return;
#pragma unroll
  for (int j = 0; j < 6; ++j) {
    pfe[i * 32 + j] = fwe[i * 6 + j];
    pfu[i * 32 + j] = fwu[i * 6 + j];
  }
#pragma unroll
  for (int j = 0; j < 36; ++j) {
    phq[i * 64 + j]   = hq[i * 36 + j];
    phsig[i * 64 + j] = hsig[i * 36 + j];
  }
#pragma unroll
  for (int j = 0; j < 49; ++j) phs[i * 64 + j] = hs[i * 49 + j];
}

// ---------------- prep: predict / diffs / clamped innovation ----------------
__global__ void prep_kernel(const float* __restrict__ x,         // [14,B]
                            const float* __restrict__ state,     // [B,7]
                            const float* __restrict__ estate,    // [B,6]
                            const float* __restrict__ lastm,     // [B,7]
                            float* __restrict__ x14,             // [B,32]
                            float* __restrict__ pred,            // [B,8]
                            float* __restrict__ dy) {            // [B,8]
  int i = blockIdx.x * 256 + threadIdx.x;
  if (i >= BATCH) return;
  float meas[7], esp[6], pr[7];
#pragma unroll
  for (int j = 0; j < 7; ++j) meas[j] = x[j * BATCH + i];
#pragma unroll
  for (int j = 0; j < 6; ++j) esp[j] = estate[i * 6 + j] + DT_C * x[(8 + j) * BATCH + i];
#pragma unroll
  for (int j = 0; j < 7; ++j) {
    float inj = (j < 3) ? esp[j] : (j < 6 ? 0.5f * esp[j] : 0.0f);
    pr[j] = state[i * 7 + j] + inj;
  }
  const float mm[7] = {0.005f, 0.005f, 0.005f, 0.1f, 0.1f, 0.1f, 0.1f};
#pragma unroll
  for (int j = 0; j < 7; ++j) {
    x14[i * 32 + j]     = meas[j] - lastm[i * 7 + j];        // obs_diff
    x14[i * 32 + 7 + j] = meas[j] - pr[j];                   // obs_innov_diff
    pred[i * 8 + j] = pr[j];
    float d = meas[j] - pr[j];
    float s = (d > 0.0f) ? 1.0f : (d < 0.0f ? -1.0f : 0.0f);
    dy[i * 8 + j] = s * fminf(fabsf(d), mm[j]);
  }
}

// ---------------- generic WMMA GEMM: C = act(A @ W^T + b) -------------------
// A f32 [B,lda] (lda >= Kp, pad cols zero). W f16 [Np,Kp] zero-padded.
// bias f32 padded to Np. Store guarded by true N only.
__global__ __launch_bounds__(128)
void gemm_wmma_kernel(const float* __restrict__ A, int lda, int Kp,
                      const _Float16* __restrict__ W,
                      const float* __restrict__ bias, int N,
                      float* __restrict__ C, int ldc, int ccol, int do_relu) {
  const int lane = threadIdx.x & 31;
  const int wid  = threadIdx.x >> 5;
  const int nTiles = (N + 15) >> 4;
  const int nt = blockIdx.y * 4 + wid;            // wave-uniform
  if (nt >= nTiles) return;                       // EXEC stays all-ones
  const int mb = blockIdx.x << 4;
  const int n  = (nt << 4) + (lane & 15);
  const int kbA = (lane < 16) ? 0 : 16;
  const int kbB = (lane < 16) ? 0 : 16;
  const float*    Arow = A + (size_t)(mb + (lane & 15)) * lda;
  const _Float16* Wrow = W + (size_t)n * Kp;
  v8f acc = {};
  for (int kt = 0; kt < Kp; kt += 32) {
    v8f a0 = *(const v8f*)(Arow + kt + kbA);
    v8f a1 = *(const v8f*)(Arow + kt + kbA + 8);
    v16h b = *(const v16h*)(Wrow + kt + kbB);
    v16h a;
#pragma unroll
    for (int e = 0; e < 8; ++e) {
      a[e]     = (_Float16)a0[e];
      a[8 + e] = (_Float16)a1[e];
    }
    acc = wmma_f16(a, b, acc);
  }
  float bs = bias[n];
  int mOff = (lane < 16) ? 0 : 8;
  if (n < N) {
#pragma unroll
    for (int v = 0; v < 8; ++v) {
      float val = acc[v] + bs;
      if (do_relu) val = fmaxf(val, 0.0f);
      C[(size_t)(mb + mOff + v) * ldc + ccol + n] = val;
    }
  }
}

// ---------------- fused: out = relu(A @ Wa^T + ba) @ Wb^T + bb --------------
// A f32 [B,lda], Kp=96. Wa f16 [Hp,96], ba f32 [Hp]. Wb f16 [16,Hp], bb f32 [16].
__global__ __launch_bounds__(128)
void fused_mlp_kernel(const float* __restrict__ A, int lda,
                      const _Float16* __restrict__ Wa, const float* __restrict__ ba,
                      int Hp,
                      const _Float16* __restrict__ Wb, const float* __restrict__ bb,
                      int N2,
                      float* __restrict__ C, int ldc, int ccol) {
  __shared__ _Float16 Hs[4][16][32];              // per-wave 16x32 hidden tile
  const int lane = threadIdx.x & 31;
  const int wid  = threadIdx.x >> 5;
  const int mtile = blockIdx.x * 4 + wid;         // wave-uniform
  if (mtile >= (BATCH >> 4)) return;
  const int mb   = mtile << 4;
  const int kbA  = (lane < 16) ? 0 : 16;
  const int kbB  = (lane < 16) ? 0 : 16;
  const int mOff = (lane < 16) ? 0 : 8;
  // preload A fragments (K=96 -> 3 chunks), reused for all hidden chunks
  v16h afr[3];
  {
    const float* Arow = A + (size_t)(mb + (lane & 15)) * lda;
#pragma unroll
    for (int c = 0; c < 3; ++c) {
      v8f a0 = *(const v8f*)(Arow + c * 32 + kbA);
      v8f a1 = *(const v8f*)(Arow + c * 32 + kbA + 8);
#pragma unroll
      for (int e = 0; e < 8; ++e) {
        afr[c][e]     = (_Float16)a0[e];
        afr[c][8 + e] = (_Float16)a1[e];
      }
    }
  }
  const int n2 = lane & 15;
  const _Float16* Wbr = Wb + (size_t)n2 * Hp;
  v8f out = {};
  for (int h0 = 0; h0 < Hp; h0 += 32) {
    // --- first GEMM: two 16-wide hidden sub-tiles -> LDS (bias+relu) ---
#pragma unroll
    for (int half = 0; half < 2; ++half) {
      const int hcol = h0 + half * 16 + (lane & 15);
      const _Float16* War = Wa + (size_t)hcol * 96;
      v8f hacc = {};
#pragma unroll
      for (int c = 0; c < 3; ++c) {
        v16h b = *(const v16h*)(War + c * 32 + kbB);
        hacc = wmma_f16(afr[c], b, hacc);
      }
      float bsv = ba[hcol];
#pragma unroll
      for (int v = 0; v < 8; ++v)
        Hs[wid][mOff + v][half * 16 + (lane & 15)] =
            (_Float16)fmaxf(hacc[v] + bsv, 0.0f);
    }
    // --- second GEMM: re-gather hidden tile as A-fragment, accumulate ---
    const int mloc = lane & 15;
    v16h a2;
    *(v8f*)&a2 = *(const v8f*)&Hs[wid][mloc][kbA];           // 16 halves (32 B)
    *(((v8f*)&a2) + 1) = *(const v8f*)&Hs[wid][mloc][kbA + 8];
    v16h b2 = *(const v16h*)(Wbr + h0 + kbB);
    out = wmma_f16(a2, b2, out);
  }
  float bs = bb[n2];
  if (n2 < N2) {
#pragma unroll
    for (int v = 0; v < 8; ++v)
      C[(size_t)(mb + mOff + v) * ldc + ccol + n2] = out[v] + bs;
  }
}

// ---------------- GRU elementwise gate combine ------------------------------
__global__ void gru_ew_kernel(const float* __restrict__ gi,
                              const float* __restrict__ gh, int ldg, int dh,
                              const float* __restrict__ h, int ldh,
                              float* __restrict__ d1, int ld1, int off1,
                              float* __restrict__ d2, int ld2, int off2) {
  int idx = blockIdx.x * 256 + threadIdx.x;
  if (idx >= BATCH * dh) return;
  int i = idx / dh, j = idx - i * dh;
  const float* gir = gi + (size_t)i * ldg;
  const float* ghr = gh + (size_t)i * ldg;
  float r = 1.0f / (1.0f + expf(-(gir[j] + ghr[j])));
  float z = 1.0f / (1.0f + expf(-(gir[dh + j] + ghr[dh + j])));
  float n = tanhf(gir[2 * dh + j] + r * ghr[2 * dh + j]);
  float o = (1.0f - z) * n + z * h[(size_t)i * ldh + j];
  d1[(size_t)i * ld1 + off1 + j] = o;
  if (d2) d2[(size_t)i * ld2 + off2 + j] = o;
}

// ---------------- final: err = K @ dy, state injection ----------------------
__global__ void final_kernel(const float* __restrict__ c70,   // [B,96] (S|FC2|FC21)
                             const float* __restrict__ pred,  // [B,8]
                             const float* __restrict__ dy,    // [B,8]
                             float* __restrict__ out) {       // [B,7]
  int i = blockIdx.x * 256 + threadIdx.x;
  if (i >= BATCH) return;
  const float* row = c70 + (size_t)i * 96;
  const float* dyr = dy + (size_t)i * 8;
  const float* prr = pred + (size_t)i * 8;
  float err[6];
#pragma unroll
  for (int r = 0; r < 3; ++r) {
    float s0 = 0.0f, s1 = 0.0f;
#pragma unroll
    for (int c = 0; c < 3; ++c) s0 += row[49 + r * 3 + c] * dyr[c];      // K00 3x3
#pragma unroll
    for (int c = 0; c < 4; ++c) s1 += row[58 + r * 4 + c] * dyr[3 + c];  // K11 3x4
    err[r] = s0; err[3 + r] = s1;
  }
#pragma unroll
  for (int j = 0; j < 7; ++j) {
    float inj = (j < 3) ? err[j] : (j < 6 ? 0.5f * err[j] : 0.0f);
    out[i * 7 + j] = prr[j] + inj;
  }
}

// ---------------- workspace layout (floats), all strides mult-of-8 ---------
static constexpr size_t O_X14  = 0;                               // [B,32]
static constexpr size_t O_PRED = O_X14  + (size_t)BATCH * 32;     // [B,8]
static constexpr size_t O_DY   = O_PRED + (size_t)BATCH * 8;      // [B,8]
static constexpr size_t O_FWE  = O_DY   + (size_t)BATCH * 8;      // [B,32]
static constexpr size_t O_FWU  = O_FWE  + (size_t)BATCH * 32;     // [B,32]
static constexpr size_t O_HQ   = O_FWU  + (size_t)BATCH * 32;     // [B,64]
static constexpr size_t O_HSG  = O_HQ   + (size_t)BATCH * 64;     // [B,64]
static constexpr size_t O_HS   = O_HSG  + (size_t)BATCH * 64;     // [B,64]
static constexpr size_t O_GI   = O_HS   + (size_t)BATCH * 64;     // [B,160]
static constexpr size_t O_GH   = O_GI   + (size_t)BATCH * 160;    // [B,160]
static constexpr size_t O_A5   = O_GH   + (size_t)BATCH * 160;    // [B,256]
static constexpr size_t O_C276 = O_A5   + (size_t)BATCH * 256;    // [B,288] Q|fc6
static constexpr size_t O_C609 = O_C276 + (size_t)BATCH * 288;    // [B,640] fc1|fc7
static constexpr size_t O_C85  = O_C609 + (size_t)BATCH * 640;    // [B,96]  Sig|S
static constexpr size_t O_C70  = O_C85  + (size_t)BATCH * 96;     // [B,96]  S|FC2|FC21
static constexpr size_t O_C72  = O_C70  + (size_t)BATCH * 96;     // [B,96]  Sig|FC3
static constexpr size_t F32_END = O_C72 + (size_t)BATCH * 96;
// f16 weight pool offsets (halfs), each [Np, Kp]
static constexpr size_t HW_Q_IH  = 0;                             // 112x256
static constexpr size_t HW_Q_HH  = HW_Q_IH  + 112 * 256;          // 112x64
static constexpr size_t HW_SG_IH = HW_Q_HH  + 112 * 64;           // 112x288
static constexpr size_t HW_SG_HH = HW_SG_IH + 112 * 288;          // 112x64
static constexpr size_t HW_S_IH  = HW_SG_HH + 112 * 64;           // 160x640
static constexpr size_t HW_S_HH  = HW_S_IH  + 160 * 640;          // 160x64
static constexpr size_t HW_FC1   = HW_S_HH  + 160 * 64;           // 64x64
static constexpr size_t HW_FC2A  = HW_FC1   + 64 * 64;            // 3424x96
static constexpr size_t HW_FC2B  = HW_FC2A  + 3424 * 96;          // 16x3424
static constexpr size_t HW_FC21A = HW_FC2B  + 16 * 3424;          // 3424x96
static constexpr size_t HW_FC21B = HW_FC21A + 3424 * 96;          // 16x3424
static constexpr size_t HW_FC3   = HW_FC21B + 16 * 3424;          // 48x96
static constexpr size_t HW_FC4   = HW_FC3   + 48 * 96;            // 48x96
static constexpr size_t HW_FC5   = HW_FC4   + 48 * 96;            // 240x32
static constexpr size_t HW_FC6   = HW_FC5   + 240 * 32;           // 240x32
static constexpr size_t HW_FC7   = HW_FC6   + 240 * 32;           // 560x32
static constexpr size_t HW_END   = HW_FC7   + 560 * 32;
// f32 bias pool offsets (floats), padded to Np
static constexpr size_t BQ_IH = 0, BQ_HH = 112, BSG_IH = 224, BSG_HH = 336;
static constexpr size_t BS_IH = 448, BS_HH = 608, BFC1 = 768, BFC2A = 832;
static constexpr size_t BFC2B = 4256, BFC21A = 4272, BFC21B = 7696;
static constexpr size_t BFC3 = 7712, BFC4 = 7760, BFC5 = 7808, BFC6 = 8048;
static constexpr size_t BFC7 = 8288;  // end 8848
// total ws ~ 137.4 MB f32 + 2.0 MB f16 + 35 KB bias

extern "C" void kernel_launch(void* const* d_in, const int* in_sizes, int n_in,
                              void* d_out, int out_size, void* d_ws, size_t ws_size,
                              hipStream_t stream) {
  (void)in_sizes; (void)n_in; (void)out_size; (void)ws_size;
  const float* x      = (const float*)d_in[0];
  const float* state  = (const float*)d_in[1];
  const float* estate = (const float*)d_in[2];
  const float* lastm  = (const float*)d_in[3];

  float* ws = (float*)d_ws;
  _Float16* hp = (_Float16*)(ws + F32_END);
  float* bp = (float*)(hp + HW_END);
  float* out = (float*)d_out;

  // ---- zero activation region (establishes all pad columns) ----
  {
    size_t n4 = F32_END / 4;
    zero_ws_kernel<<<(unsigned)((n4 + 255) / 256), 256, 0, stream>>>((float4*)ws, n4);
  }

  // ---- weights -> padded f16 pool; biases -> padded f32 pool ----
  struct { int wi; size_t woff; int N, K, Kp; int bi; size_t boff; int Np; } wc[] = {
    {9,  HW_Q_IH,  108, 240, 256, 11, BQ_IH,  112},
    {10, HW_Q_HH,  108, 36,  64,  12, BQ_HH,  112},
    {13, HW_SG_IH, 108, 276, 288, 15, BSG_IH, 112},
    {14, HW_SG_HH, 108, 36,  64,  16, BSG_HH, 112},
    {17, HW_S_IH,  147, 609, 640, 19, BS_IH,  160},
    {18, HW_S_HH,  147, 49,  64,  20, BS_HH,  160},
    {21, HW_FC1,   49,  36,  64,  22, BFC1,   64},
    {23, HW_FC2A,  3400, 85, 96,  24, BFC2A,  3424},
    {25, HW_FC2B,  9, 3400, 3424, 26, BFC2B,  16},
    {27, HW_FC21A, 3400, 85, 96,  28, BFC21A, 3424},
    {29, HW_FC21B, 12, 3400, 3424, 30, BFC21B, 16},
    {31, HW_FC3,   36,  70,  96,  32, BFC3,   48},
    {33, HW_FC4,   36,  72,  96,  34, BFC4,   48},
    {35, HW_FC5,   240, 6,   32,  36, BFC5,   240},
    {37, HW_FC6,   240, 6,   32,  38, BFC6,   240},
    {39, HW_FC7,   560, 14,  32,  40, BFC7,   560},
  };
  for (auto& w : wc) {
    int total = w.Np * w.Kp;
    cvt_pad_w<<<(total + 255) / 256, 256, 0, stream>>>(
        (const float*)d_in[w.wi], hp + w.woff, w.N, w.K, w.Kp, total);
    pad_bias<<<(w.Np + 255) / 256, 256, 0, stream>>>(
        (const float*)d_in[w.bi], bp + w.boff, w.N, w.Np);
  }

  // ---- pack small inputs into padded rows; prep kernel ----
  pack_inputs<<<BATCH / 256, 256, 0, stream>>>(
      (const float*)d_in[4], (const float*)d_in[5],
      (const float*)d_in[6], (const float*)d_in[7], (const float*)d_in[8],
      ws + O_FWE, ws + O_FWU, ws + O_HQ, ws + O_HSG, ws + O_HS);
  prep_kernel<<<BATCH / 256, 256, 0, stream>>>(
      x, state, estate, lastm, ws + O_X14, ws + O_PRED, ws + O_DY);

  auto gemm = [&](const float* A, int lda, int Kp, const _Float16* W,
                  const float* bias, int N, float* C, int ldc, int ccol, int relu) {
    int nt = (N + 15) / 16;
    dim3 g(BATCH / 16, (nt + 3) / 4);
    gemm_wmma_kernel<<<g, 128, 0, stream>>>(A, lda, Kp, W, bias, N, C, ldc, ccol, relu);
  };

  // ---- fc5 / fc6 ----
  gemm(ws + O_FWE, 32, 32, hp + HW_FC5, bp + BFC5, 240, ws + O_A5, 256, 0, 1);
  gemm(ws + O_FWU, 32, 32, hp + HW_FC6, bp + BFC6, 240, ws + O_C276, 288, 36, 1);
  // ---- gru_q ----
  gemm(ws + O_A5, 256, 256, hp + HW_Q_IH, bp + BQ_IH, 108, ws + O_GI, 160, 0, 0);
  gemm(ws + O_HQ, 64, 64,   hp + HW_Q_HH, bp + BQ_HH, 108, ws + O_GH, 160, 0, 0);
  gru_ew_kernel<<<(BATCH * 36 + 255) / 256, 256, 0, stream>>>(
      ws + O_GI, ws + O_GH, 160, 36, ws + O_HQ, 64,
      ws + O_C276, 288, 0, (float*)nullptr, 0, 0);
  // ---- gru_sigma ----
  gemm(ws + O_C276, 288, 288, hp + HW_SG_IH, bp + BSG_IH, 108, ws + O_GI, 160, 0, 0);
  gemm(ws + O_HSG, 64, 64,    hp + HW_SG_HH, bp + BSG_HH, 108, ws + O_GH, 160, 0, 0);
  gru_ew_kernel<<<(BATCH * 36 + 255) / 256, 256, 0, stream>>>(
      ws + O_GI, ws + O_GH, 160, 36, ws + O_HSG, 64,
      ws + O_C85, 96, 0, ws + O_C72, 96, 0);
  // ---- fc1 / fc7 ----
  gemm(ws + O_C85, 96, 64, hp + HW_FC1, bp + BFC1, 49, ws + O_C609, 640, 0, 1);
  gemm(ws + O_X14, 32, 32, hp + HW_FC7, bp + BFC7, 560, ws + O_C609, 640, 49, 1);
  // ---- gru_s ----
  gemm(ws + O_C609, 640, 640, hp + HW_S_IH, bp + BS_IH, 147, ws + O_GI, 160, 0, 0);
  gemm(ws + O_HS, 64, 64,     hp + HW_S_HH, bp + BS_HH, 147, ws + O_GH, 160, 0, 0);
  gru_ew_kernel<<<(BATCH * 49 + 255) / 256, 256, 0, stream>>>(
      ws + O_GI, ws + O_GH, 160, 49, ws + O_HS, 64,
      ws + O_C85, 96, 36, ws + O_C70, 96, 0);
  // ---- fused fc2a->fc2b / fc21a->fc21b (hidden 3400 never hits HBM) ----
  fused_mlp_kernel<<<BATCH / 64, 128, 0, stream>>>(
      ws + O_C85, 96, hp + HW_FC2A, bp + BFC2A, 3424,
      hp + HW_FC2B, bp + BFC2B, 9, ws + O_C70, 96, 49);
  fused_mlp_kernel<<<BATCH / 64, 128, 0, stream>>>(
      ws + O_C85, 96, hp + HW_FC21A, bp + BFC21A, 3424,
      hp + HW_FC21B, bp + BFC21B, 12, ws + O_C70, 96, 58);
  // ---- fc3 / fc4 ----
  gemm(ws + O_C70, 96, 96, hp + HW_FC3, bp + BFC3, 36, ws + O_C72, 96, 36, 1);
  gemm(ws + O_C72, 96, 96, hp + HW_FC4, bp + BFC4, 36,
       out + (size_t)BATCH * 7, 36, 0, 1);           // out_FC4 -> second output
  // ---- Kalman gain apply + state injection -> first output ----
  final_kernel<<<BATCH / 256, 256, 0, stream>>>(
      ws + O_C70, ws + O_PRED, ws + O_DY, out);
}
